// ThreeBodySpringMassGraphModel_21818433863949
// MI455X (gfx1250) — matrix operand order repaired and checked by reading, not compiled, not run on hardware
//
#include <hip/hip_runtime.h>

// ---------------------------------------------------------------------------
// MI455X (gfx1250, wave32) GNN forward: all MLPs via v_wmma_f32_16x16x32_f16,
// weights staged in LDS (bank-conflict-padded), B fragments prefetched into a
// register array per K-step so ds_loads clause ahead of the WMMA burst.
// ---------------------------------------------------------------------------

typedef __attribute__((ext_vector_type(16))) _Float16 v16h;
typedef __attribute__((ext_vector_type(8)))  _Float16 v8h;
typedef __attribute__((ext_vector_type(8)))  float    v8f;

#define HIDDEN 128
#define NB 8
#define NN 128
#define NS 2
#define NROWS (NB*NN*NS)      /* 2048  node rows  */
#define EROWS (NB*NN*NN*NS)   /* 262144 edge rows */
#define LPAD  8               /* LDS row pad (halves) -> lanes staggered 4 words */

// ---- A-fragment loader: 16x32 f16 A layout (lane half h: K=8h..8h+7 then
// K=16+8h..16+8h+7; two contiguous 16B chunks per lane) ---------------------
__device__ __forceinline__ v16h load_a_frag(const _Float16* p) {
  v8h lo = *(const v8h*)(p);
  v8h hi = *(const v8h*)(p + 16);
  return __builtin_shufflevector(lo, hi, 0,1,2,3,4,5,6,7,8,9,10,11,12,13,14,15);
}

// ---- B-fragment loader from LDS: two 8-half chunks at p and p+16, 16B-aligned ---------
__device__ __forceinline__ v16h load_b_frag(const _Float16* p) {
  v8h lo = *(const v8h*)(p);
  v8h hi = *(const v8h*)(p + 16);
  return __builtin_shufflevector(lo, hi, 0,1,2,3,4,5,6,7,8,9,10,11,12,13,14,15);
}

__device__ __forceinline__ v8f wmma16(v16h a, v16h b, v8f c) {
  return __builtin_amdgcn_wmma_f32_16x16x32_f16(false, a, false, b,
                                                (short)0, c, false, false);
}

// Cooperative copy of Wt[128][K] (f16, N-major) into padded LDS [128][K+LPAD].
template<int K>
__device__ __forceinline__ void stage_weights(const _Float16* __restrict__ Wt,
                                              _Float16* __restrict__ wlds) {
  constexpr int CPR = K / 8;                 // 16B chunks per row
  for (int i = threadIdx.x; i < 128 * CPR; i += 256) {
    int n = i / CPR, c = i - n * CPR;
    *(uint4*)(wlds + n * (K + LPAD) + c * 8) = ((const uint4*)(Wt + n * K))[c];
  }
  __syncthreads();
}

// ---------------------------------------------------------------------------
// Generic GEMM: Y[R,128] = act(X[R,K] @ W[K,128] + bias)
// X f16 row-major (stride K), Wt f16 N-major [128][K], Y f16 [R,128].
// Block = 256 threads (8 waves); wave computes 32 rows x 128 cols.
// ---------------------------------------------------------------------------
template<int K>
__global__ __launch_bounds__(256) void gemm128_t(
    const _Float16* __restrict__ X,
    const _Float16* __restrict__ Wt, const float* __restrict__ bias,
    _Float16* __restrict__ Y, int relu)
{
  extern __shared__ _Float16 wlds[];
  stage_weights<K>(Wt, wlds);

  const int lane = threadIdx.x & 31;
  const int wv   = threadIdx.x >> 5;
  const int l16  = lane & 15;
  const int h    = lane >> 4;
  const long rowBase = (long)blockIdx.x * 256 + wv * 32;
  constexpr int KP = K + LPAD;

  v8f acc[2][8];
#pragma unroll
  for (int nt = 0; nt < 8; ++nt) {
    float bv = bias[nt*16 + l16];
    v8f c = {bv,bv,bv,bv,bv,bv,bv,bv};
    acc[0][nt] = c; acc[1][nt] = c;
  }

  const _Float16* xa0 = X + (rowBase +  0 + l16) * K + 8*h;
  const _Float16* xa1 = X + (rowBase + 16 + l16) * K + 8*h;
  // Per-nt LDS base pointers hoisted out of the K loop: remaining DS offset is
  // k0*2 bytes <= 768, always within the 16-bit ds immediate.
  const _Float16* wp[8];
#pragma unroll
  for (int nt = 0; nt < 8; ++nt)
    wp[nt] = wlds + (nt*16 + l16) * KP + 8*h;

#pragma unroll
  for (int k0 = 0; k0 < K; k0 += 32) {
    v16h a0 = load_a_frag(xa0 + k0);
    v16h a1 = load_a_frag(xa1 + k0);
    v16h bfr[8];
#pragma unroll
    for (int nt = 0; nt < 8; ++nt)
      bfr[nt] = load_b_frag(wp[nt] + k0);
#pragma unroll
    for (int nt = 0; nt < 8; ++nt) {
      acc[0][nt] = wmma16(a0, bfr[nt], acc[0][nt]);
      acc[1][nt] = wmma16(a1, bfr[nt], acc[1][nt]);
    }
  }

#pragma unroll
  for (int mt = 0; mt < 2; ++mt) {
    const long r0 = rowBase + mt*16 + 8*h;
#pragma unroll
    for (int nt = 0; nt < 8; ++nt) {
      const int col = nt*16 + l16;
      v8f c = acc[mt][nt];
#pragma unroll
      for (int i = 0; i < 8; ++i) {
        float v = c[i];
        if (relu) v = fmaxf(v, 0.f);
        Y[(r0 + i)*HIDDEN + col] = (_Float16)v;
      }
    }
  }
}

// ---------------------------------------------------------------------------
// Fused-gather message GEMM: K = 384 = [h_n[sender] | h_n[receiver] | h_e].
// Edge row r: e = r>>1, s = r&1, b = e>>14, recv = (e>>7)&127, send = e&127.
// Avoids materializing the 192MB concat buffer; weights staged in LDS.
// ---------------------------------------------------------------------------
__global__ __launch_bounds__(256) void msg_gemm(
    const _Float16* __restrict__ hn, const _Float16* __restrict__ he,
    const _Float16* __restrict__ Wt, const float* __restrict__ bias,
    _Float16* __restrict__ Y)
{
  constexpr int K  = 384;
  constexpr int KP = K + LPAD;
  extern __shared__ _Float16 wlds[];
  stage_weights<K>(Wt, wlds);

  const int lane = threadIdx.x & 31;
  const int wv   = threadIdx.x >> 5;
  const int l16  = lane & 15;
  const int h    = lane >> 4;
  const int rowBase = blockIdx.x * 256 + wv * 32;

  v8f acc[2][8];
#pragma unroll
  for (int nt = 0; nt < 8; ++nt) {
    float bv = bias[nt*16 + l16];
    v8f c = {bv,bv,bv,bv,bv,bv,bv,bv};
    acc[0][nt] = c; acc[1][nt] = c;
  }

  const _Float16* P[2][3];
#pragma unroll
  for (int mt = 0; mt < 2; ++mt) {
    int row  = rowBase + mt*16 + l16;
    int e    = row >> 1, s = row & 1;
    int b    = e >> 14;
    int rem  = e & 16383;
    int recv = rem >> 7;
    int send = rem & 127;
    P[mt][0] = hn + ((((b<<7) + send)<<1) + s) * HIDDEN + 8*h;
    P[mt][1] = hn + ((((b<<7) + recv)<<1) + s) * HIDDEN + 8*h;
    P[mt][2] = he + (long)row * HIDDEN + 8*h;
  }

  const _Float16* wp[8];
#pragma unroll
  for (int nt = 0; nt < 8; ++nt)
    wp[nt] = wlds + (nt*16 + l16) * KP + 8*h;

#pragma unroll
  for (int seg = 0; seg < 3; ++seg) {
#pragma unroll
    for (int lk = 0; lk < 128; lk += 32) {
      v16h a0 = load_a_frag(P[0][seg] + lk);
      v16h a1 = load_a_frag(P[1][seg] + lk);
      const int k0 = seg*128 + lk;
      v16h bfr[8];
#pragma unroll
      for (int nt = 0; nt < 8; ++nt)
        bfr[nt] = load_b_frag(wp[nt] + k0);
#pragma unroll
      for (int nt = 0; nt < 8; ++nt) {
        acc[0][nt] = wmma16(a0, bfr[nt], acc[0][nt]);
        acc[1][nt] = wmma16(a1, bfr[nt], acc[1][nt]);
      }
    }
  }

#pragma unroll
  for (int mt = 0; mt < 2; ++mt) {
    const long r0 = (long)rowBase + mt*16 + 8*h;
#pragma unroll
    for (int nt = 0; nt < 8; ++nt) {
      const int col = nt*16 + l16;
      v8f c = acc[mt][nt];
#pragma unroll
      for (int i = 0; i < 8; ++i) {
        float v = fmaxf(c[i], 0.f);   // first MLP layer: always ReLU
        Y[(r0 + i)*HIDDEN + col] = (_Float16)v;
      }
    }
  }
}

// ---------------------------------------------------------------------------
// Segment-sum (edges for a receiver are contiguous) + concat into [h_n | agg]
// ---------------------------------------------------------------------------
__global__ void agg_concat(const _Float16* __restrict__ he,
                           const _Float16* __restrict__ hn,
                           _Float16* __restrict__ nodecat)
{
  const int nrow = blockIdx.x;      // 0..2047
  const int t    = threadIdx.x;     // 0..127
  const int b    = nrow >> 8;
  const int rem  = nrow & 255;
  const int r    = rem >> 1;
  const int s    = rem & 1;
  const long ebase = ((long)b << 14) + ((long)r << 7);
  float sum = 0.f;
  for (int j = 0; j < NN; ++j)
    sum += (float)he[(((ebase + j) << 1) + s) * HIDDEN + t];
  nodecat[(long)nrow*256 + t]        = hn[(long)nrow*HIDDEN + t];
  nodecat[(long)nrow*256 + 128 + t]  = (_Float16)sum;
}

// ---------------------------------------------------------------------------
// Input builders (zero-padded f16 feature matrices, K padded to 32)
// ---------------------------------------------------------------------------
__global__ void build_nodeX(const float* __restrict__ dq1, const float* __restrict__ dq2,
                            const float* __restrict__ dp1, const float* __restrict__ dp2,
                            const float* __restrict__ m, _Float16* __restrict__ nodeX)
{
  int idx = blockIdx.x*256 + threadIdx.x;
  if (idx >= NROWS*32) return;
  int row = idx >> 5, c = idx & 31;
  float v = 0.f;
  if      (c == 0) v = dq1[row];
  else if (c == 1) v = dq2[row];
  else if (c == 2) v = dp1[row];
  else if (c == 3) v = dp2[row];
  else if (c == 4) v = m[row >> 1];
  nodeX[idx] = (_Float16)v;
}

__global__ void build_edgeX(const float* __restrict__ length, const float* __restrict__ kk,
                            _Float16* __restrict__ edgeX)
{
  long idx = (long)blockIdx.x*256 + threadIdx.x;
  if (idx >= (long)EROWS*32) return;
  long row = idx >> 5; int c = (int)(idx & 31);
  long e = row >> 1;
  float v = 0.f;
  if      (c == 0) v = length[e];
  else if (c == 1) v = kk[e];
  edgeX[idx] = (_Float16)v;
}

// Transpose/pad weight [K,128] fp32 -> Wt f16 [128][Kpad] (N-major)
__global__ void wtrans(const float* __restrict__ w, _Float16* __restrict__ wt,
                       int K, int Kpad)
{
  int idx = blockIdx.x*256 + threadIdx.x;
  if (idx >= 128*Kpad) return;
  int n = idx / Kpad, k2 = idx - n*Kpad;
  wt[idx] = (k2 < K) ? (_Float16)w[k2*128 + n] : (_Float16)0.f;
}

// Final decode layer: [2048,128] f16 @ [128,2] fp32 + b -> d_out (out0 | out1)
__global__ void dec_final(const _Float16* __restrict__ x, const float* __restrict__ w,
                          const float* __restrict__ bias, float* __restrict__ out)
{
  int row = blockIdx.x*256 + threadIdx.x;
  if (row >= NROWS) return;
  float o0 = bias[0], o1 = bias[1];
  for (int k2 = 0; k2 < HIDDEN; ++k2) {
    float xv = (float)x[(long)row*HIDDEN + k2];
    o0 += xv * w[k2*2 + 0];
    o1 += xv * w[k2*2 + 1];
  }
  out[row]         = o0;
  out[NROWS + row] = o1;
}

// ---------------------------------------------------------------------------
extern "C" void kernel_launch(void* const* d_in, const int* in_sizes, int n_in,
                              void* d_out, int out_size, void* d_ws, size_t ws_size,
                              hipStream_t stream)
{
  (void)in_sizes; (void)n_in; (void)out_size;
  // setup_inputs() insertion order:
  const float* dq1    = (const float*)d_in[0];
  const float* dq2    = (const float*)d_in[1];
  const float* dp1    = (const float*)d_in[2];
  const float* dp2    = (const float*)d_in[3];
  const float* m      = (const float*)d_in[4];
  const float* length = (const float*)d_in[7];
  const float* kmat   = (const float*)d_in[8];
  // params pytree leaves (dict keys sorted: dec_node, enc_edge, enc_node, mp)
  const float* dec_w0 = (const float*)d_in[9];
  const float* dec_b0 = (const float*)d_in[10];
  const float* dec_w1 = (const float*)d_in[11];
  const float* dec_b1 = (const float*)d_in[12];
  const float* ee_w0  = (const float*)d_in[13];
  const float* ee_b0  = (const float*)d_in[14];
  const float* ee_w1  = (const float*)d_in[15];
  const float* ee_b1  = (const float*)d_in[16];
  const float* en_w0  = (const float*)d_in[17];
  const float* en_b0  = (const float*)d_in[18];
  const float* en_w1  = (const float*)d_in[19];
  const float* en_b1  = (const float*)d_in[20];
  const float* mp_e_w0[2], *mp_e_b0[2], *mp_e_w1[2], *mp_e_b1[2];
  const float* mp_n_w0[2], *mp_n_b0[2], *mp_n_w1[2], *mp_n_b1[2];
  for (int l = 0; l < 2; ++l) {
    int base = 21 + l*8;
    mp_e_w0[l] = (const float*)d_in[base+0];
    mp_e_b0[l] = (const float*)d_in[base+1];
    mp_e_w1[l] = (const float*)d_in[base+2];
    mp_e_b1[l] = (const float*)d_in[base+3];
    mp_n_w0[l] = (const float*)d_in[base+4];
    mp_n_b0[l] = (const float*)d_in[base+5];
    mp_n_w1[l] = (const float*)d_in[base+6];
    mp_n_b1[l] = (const float*)d_in[base+7];
  }

  // ---- workspace bump allocator (256B aligned) ----
  char* wsp = (char*)d_ws;
  size_t off = 0;
  auto halloc = [&](size_t nhalves) -> _Float16* {
    _Float16* p = (_Float16*)(wsp + off);
    off += ((nhalves*sizeof(_Float16) + 255) & ~(size_t)255);
    return p;
  };
  _Float16* he      = halloc((size_t)EROWS*HIDDEN);   // 64 MB
  _Float16* hmid    = halloc((size_t)EROWS*HIDDEN);   // 64 MB
  _Float16* edgeX   = halloc((size_t)EROWS*32);       // 16 MB
  _Float16* nodeX   = halloc((size_t)NROWS*32);
  _Float16* hnA     = halloc((size_t)NROWS*HIDDEN);
  _Float16* hnB     = halloc((size_t)NROWS*HIDDEN);
  _Float16* hnT     = halloc((size_t)NROWS*HIDDEN);
  _Float16* nodecat = halloc((size_t)NROWS*256);
  _Float16* wt_en0  = halloc(128*32);
  _Float16* wt_en1  = halloc(128*128);
  _Float16* wt_ee0  = halloc(128*32);
  _Float16* wt_ee1  = halloc(128*128);
  _Float16* wt_me0[2], *wt_me1[2], *wt_mn0[2], *wt_mn1[2];
  for (int l = 0; l < 2; ++l) {
    wt_me0[l] = halloc(128*384);
    wt_me1[l] = halloc(128*128);
    wt_mn0[l] = halloc(128*256);
    wt_mn1[l] = halloc(128*128);
  }
  _Float16* wt_dec0 = halloc(128*128);
  if (off > ws_size) return;   // workspace too small: do nothing (deterministic)

  // ---- weight transposes (fp32 -> f16, N-major, K zero-padded) ----
  auto wtl = [&](const float* w, _Float16* wt, int K, int Kpad) {
    int total = 128*Kpad;
    wtrans<<<(total+255)/256, 256, 0, stream>>>(w, wt, K, Kpad);
  };
  wtl(en_w0, wt_en0,   5,  32);
  wtl(en_w1, wt_en1, 128, 128);
  wtl(ee_w0, wt_ee0,   2,  32);
  wtl(ee_w1, wt_ee1, 128, 128);
  for (int l = 0; l < 2; ++l) {
    wtl(mp_e_w0[l], wt_me0[l], 384, 384);
    wtl(mp_e_w1[l], wt_me1[l], 128, 128);
    wtl(mp_n_w0[l], wt_mn0[l], 256, 256);
    wtl(mp_n_w1[l], wt_mn1[l], 128, 128);
  }
  wtl(dec_w0, wt_dec0, 128, 128);

  // ---- feature builders ----
  build_nodeX<<<(NROWS*32+255)/256, 256, 0, stream>>>(dq1, dq2, dp1, dp2, m, nodeX);
  build_edgeX<<<(int)(((long)EROWS*32+255)/256), 256, 0, stream>>>(length, kmat, edgeX);

  const int GN = NROWS/256;   // 8
  const int GE = EROWS/256;   // 1024
  auto shb = [](int K){ return (size_t)(K + LPAD) * 128 * sizeof(_Float16); };

  // ---- encoders ----
  gemm128_t< 32><<<GN, 256, shb( 32), stream>>>(nodeX, wt_en0, en_b0, hnT, 1);
  gemm128_t<128><<<GN, 256, shb(128), stream>>>(hnT,   wt_en1, en_b1, hnA, 0);
  gemm128_t< 32><<<GE, 256, shb( 32), stream>>>(edgeX, wt_ee0, ee_b0, hmid, 1);
  gemm128_t<128><<<GE, 256, shb(128), stream>>>(hmid,  wt_ee1, ee_b1, he,   0);

  // ---- message passing ----
  _Float16* hn = hnA;
  _Float16* hn_next = hnB;
  for (int l = 0; l < 2; ++l) {
    msg_gemm<<<GE, 256, shb(384), stream>>>(hn, he, wt_me0[l], mp_e_b0[l], hmid);
    gemm128_t<128><<<GE, 256, shb(128), stream>>>(hmid, wt_me1[l], mp_e_b1[l], he, 0);
    agg_concat<<<NROWS, 128, 0, stream>>>(he, hn, nodecat);
    gemm128_t<256><<<GN, 256, shb(256), stream>>>(nodecat, wt_mn0[l], mp_n_b0[l], hnT, 1);
    gemm128_t<128><<<GN, 256, shb(128), stream>>>(hnT,     wt_mn1[l], mp_n_b1[l], hn_next, 0);
    _Float16* tmp = hn; hn = hn_next; hn_next = tmp;
  }

  // ---- decoder ----
  gemm128_t<128><<<GN, 256, shb(128), stream>>>(hn, wt_dec0, dec_b0, hnT, 1);
  dec_final<<<(NROWS+255)/256, 256, 0, stream>>>(hnT, dec_w1, dec_b1, (float*)d_out);
}
